// HardNegativeNTXentLoss_26989574488179
// MI455X (gfx1250) — matrix operand order repaired and checked by reading, not compiled, not run on hardware
//
#include <hip/hip_runtime.h>
#include <hip/hip_bf16.h>
#include <stdint.h>

typedef __attribute__((ext_vector_type(16))) _Float16 v16h;
typedef __attribute__((ext_vector_type(8)))  _Float16 v8h;
typedef __attribute__((ext_vector_type(4)))  _Float16 v4h;
typedef __attribute__((ext_vector_type(8)))  float    v8f;

#define TWO_B   8192
#define B_ROWS  4096
#define DIM     128
#define K_SEL   4095              // int((2B-1)*0.5)
#define INV_T   14.2857142857142857f  // 1/0.07

// ---------------------------------------------------------------------------
// Kernel 1: L2-normalize rows of z1,z2 (f32) -> Zh (f16) [8192 x 128]
// One wave32 per row; each lane handles 4 contiguous floats.
// ---------------------------------------------------------------------------
__global__ __launch_bounds__(256)
void hn_normalize_kernel(const float* __restrict__ z1,
                         const float* __restrict__ z2,
                         _Float16* __restrict__ zh) {
    const int row  = (blockIdx.x * blockDim.x + threadIdx.x) >> 5;
    const int lane = threadIdx.x & 31;
    if (row >= TWO_B) return;
    const float* src = (row < B_ROWS) ? (z1 + (size_t)row * DIM)
                                      : (z2 + (size_t)(row - B_ROWS) * DIM);
    float4 v = *(const float4*)(src + lane * 4);
    float ss = v.x * v.x + v.y * v.y + v.z * v.z + v.w * v.w;
    // wave32 butterfly reduction
    #pragma unroll
    for (int off = 16; off > 0; off >>= 1)
        ss += __shfl_xor(ss, off, 32);
    const float inv = 1.0f / fmaxf(sqrtf(ss), 1e-12f);
    v4h h;
    h.x = (_Float16)(v.x * inv);
    h.y = (_Float16)(v.y * inv);
    h.z = (_Float16)(v.z * inv);
    h.w = (_Float16)(v.w * inv);
    *(v4h*)(zh + (size_t)row * DIM + lane * 4) = h;
}

// ---------------------------------------------------------------------------
// Kernel 2: sim = (Zh * Zh^T) * (1/T), f32 output [8192 x 8192].
// Block = 256 threads = 8 waves; block computes a 128x128 macro-tile.
// Each wave computes a 16x128 strip: A regs (16x128, 4 K-slices) loaded once,
// 8 column tiles x 4 x v_wmma_f32_16x16x32_f16.
//
// Register layouts used here (A/B share this K order; C/D as in CDNA5 ISA):
//   A 16x32 f16 : lane L -> M = L&15; K=[k0+hi*16, +16) contiguous
//   B 32x16 f16 : lane L -> N = L&15; K=[k0+hi*16, +16) contiguous (B = Z^T)
//   C/D 16x16 f32: VGPR r -> row r + hi*8, col L&15           (hi = L>>4)
// ---------------------------------------------------------------------------
__global__ __launch_bounds__(256)
void hn_sim_wmma_kernel(const _Float16* __restrict__ zh,
                        float* __restrict__ sim) {
    const int wid  = threadIdx.x >> 5;
    const int lane = threadIdx.x & 31;
    const int mrow = lane & 15;
    const int hi   = lane >> 4;

    const int m0 = blockIdx.x * 128 + wid * 16;   // row tile base (this wave)
    const int n0 = blockIdx.y * 128;              // col macro-tile base

    // Load A: rows m0..m0+15, all K=128, as 4 K-slices of v16h.
    union { v16h v; v8h h[2]; } a[4];
    const _Float16* arow = zh + (size_t)(m0 + mrow) * DIM;
    #pragma unroll
    for (int ks = 0; ks < 4; ++ks) {
        const int k0 = ks * 32;
        a[ks].h[0] = *(const v8h*)(arow + k0 + hi * 16);
        a[ks].h[1] = *(const v8h*)(arow + k0 + hi * 16 + 8);
    }

    #pragma unroll
    for (int ct = 0; ct < 8; ++ct) {
        const int tn = n0 + ct * 16;
        const _Float16* brow = zh + (size_t)(tn + mrow) * DIM;
        v8f c = {};
        #pragma unroll
        for (int ks = 0; ks < 4; ++ks) {
            v16h b = *(const v16h*)(brow + ks * 32 + hi * 16);
            c = __builtin_amdgcn_wmma_f32_16x16x32_f16(
                    /*neg_a=*/false, a[ks].v, /*neg_b=*/false, b,
                    /*c_mod=*/(short)0, c, /*reuse_a=*/false, /*reuse_b=*/false);
        }
        const int col   = tn + mrow;
        const int rbase = m0 + hi * 8;
        #pragma unroll
        for (int r = 0; r < 8; ++r)
            sim[(size_t)(rbase + r) * TWO_B + col] = c[r] * INV_T;
    }
}

// ---------------------------------------------------------------------------
// Kernel 3: per-row exact top-K_SEL selection (4-pass radix select on
// monotone-mapped float bits, all in LDS) + sum(exp(topk)) + row loss.
// One 256-thread block per row. Deterministic (fixed LDS tree reduction).
// key mapping (descending order == descending uint):
//   enc(f) = bits(f) sign? ~bits : bits|0x80000000 ; diag forced to key 0.
// ---------------------------------------------------------------------------
__global__ __launch_bounds__(256)
void hn_loss_kernel(const float* __restrict__ sim,
                    float* __restrict__ lossRow) {
    __shared__ uint32_t keys[TWO_B];   // 32 KB
    __shared__ uint32_t hist[256];     // 1 KB
    __shared__ float    fpart[256];    // 1 KB
    __shared__ uint32_t s_prefix, s_mask, s_kk, s_greater;
    __shared__ float    s_pos;

    const int row = blockIdx.x;
    const int t   = threadIdx.x;
    const int posCol = row ^ B_ROWS;   // (row + B) mod 2B, B = 4096 = 2^12
    const float* srow = sim + (size_t)row * TWO_B;

    if (t == 0) { s_prefix = 0u; s_mask = 0u; s_kk = K_SEL; s_greater = 0u; s_pos = 0.f; }
    __syncthreads();

    for (int j = t; j < TWO_B; j += 256) {
        const float v = srow[j];
        if (j == posCol) s_pos = v;            // single writer
        uint32_t key;
        if (j == row) {
            key = 0u;                          // masked diagonal: smallest key
        } else {
            const uint32_t b = __float_as_uint(v);
            key = (b & 0x80000000u) ? ~b : (b | 0x80000000u);
        }
        keys[j] = key;
    }
    __syncthreads();

    // 4 x 8-bit radix-select passes, MSB first, descending rank s_kk (1-based).
    for (int pass = 0; pass < 4; ++pass) {
        hist[t] = 0u;
        __syncthreads();
        const uint32_t mask   = s_mask;
        const uint32_t prefix = s_prefix;
        const int shift = 24 - 8 * pass;
        for (int j = t; j < TWO_B; j += 256) {
            const uint32_t key = keys[j];
            if ((key & mask) == prefix)
                atomicAdd(&hist[(key >> shift) & 0xFFu], 1u);
        }
        __syncthreads();
        if (t == 0) {
            uint32_t kk = s_kk, cum = 0u;
            int b;
            for (b = 255; b > 0; --b) {
                const uint32_t c = hist[b];
                if (cum + c >= kk) break;
                cum += c;
            }
            s_greater += cum;
            s_kk = kk - cum;
            s_prefix = prefix | ((uint32_t)b << shift);
            s_mask   = mask | (0xFFu << shift);
        }
        __syncthreads();
    }

    const uint32_t tkey = s_prefix;            // key of the K_SEL-th largest
    const uint32_t ties = s_kk;                // how many == tkey to include

    // Sum exp over strictly-greater keys.
    float local = 0.f;
    for (int j = t; j < TWO_B; j += 256) {
        const uint32_t key = keys[j];
        if (key > tkey) {
            const uint32_t b = (key & 0x80000000u) ? (key ^ 0x80000000u) : ~key;
            local += __expf(__uint_as_float(b));
        }
    }
    fpart[t] = local;
    __syncthreads();
    #pragma unroll
    for (int s = 128; s > 0; s >>= 1) {
        if (t < s) fpart[t] += fpart[t + s];
        __syncthreads();
    }

    if (t == 0) {
        const uint32_t tb = (tkey & 0x80000000u) ? (tkey ^ 0x80000000u) : ~tkey;
        const float tval = __uint_as_float(tb);
        const float S = fpart[0] + (float)ties * __expf(tval);
        const float pos = s_pos;
        // -log(e^pos / (e^pos + S)) = log(e^pos + S) - pos
        lossRow[row] = __logf(__expf(pos) + S) - pos;
    }
}

// ---------------------------------------------------------------------------
// Kernel 4: deterministic mean over 8192 row losses.
// ---------------------------------------------------------------------------
__global__ __launch_bounds__(256)
void hn_reduce_mean_kernel(const float* __restrict__ lossRow,
                           float* __restrict__ out) {
    __shared__ float p[256];
    const int t = threadIdx.x;
    float s = 0.f;
    for (int j = t; j < TWO_B; j += 256) s += lossRow[j];
    p[t] = s;
    __syncthreads();
    #pragma unroll
    for (int k = 128; k > 0; k >>= 1) {
        if (t < k) p[t] += p[t + k];
        __syncthreads();
    }
    if (t == 0) out[0] = p[0] * (1.0f / (float)TWO_B);
}

// ---------------------------------------------------------------------------
// Workspace layout (requires ws_size >= 2MB + 256MB + 32KB):
//   [0,          2MB)         : Zh  f16 [8192 x 128]
//   [2MB,        2MB + 256MB) : sim f32 [8192 x 8192]
//   [2MB+256MB,  +32KB)       : per-row loss f32 [8192]
// ---------------------------------------------------------------------------
extern "C" void kernel_launch(void* const* d_in, const int* in_sizes, int n_in,
                              void* d_out, int out_size, void* d_ws, size_t ws_size,
                              hipStream_t stream) {
    const float* z1 = (const float*)d_in[0];
    const float* z2 = (const float*)d_in[1];
    float* out = (float*)d_out;

    _Float16* zh  = (_Float16*)d_ws;
    float* sim    = (float*)((char*)d_ws + ((size_t)2 << 20));
    float* lossRow = (float*)((char*)d_ws + ((size_t)2 << 20) + (size_t)TWO_B * TWO_B * sizeof(float));

    // 1) normalize: 8192 rows, one wave32 per row, 8 rows per 256-thread block
    hn_normalize_kernel<<<TWO_B / 8, 256, 0, stream>>>(z1, z2, zh);

    // 2) WMMA GEMM: 64x64 macro-tiles of 128x128
    hn_sim_wmma_kernel<<<dim3(64, 64), 256, 0, stream>>>(zh, sim);

    // 3) per-row radix-select + exp-sum + loss
    hn_loss_kernel<<<TWO_B, 256, 0, stream>>>(sim, lossRow);

    // 4) deterministic mean
    hn_reduce_mean_kernel<<<1, 256, 0, stream>>>(lossRow, out);
}
